// Model_89292370084462
// MI455X (gfx1250) — matrix operand order repaired and checked by reading, not compiled, not run on hardware
//
#include <hip/hip_runtime.h>
#include <hip/hip_bf16.h>
#include <math.h>

// ---------------- problem dimensions ----------------
#define BATCH     32
#define SEQ_LEN   512
#define N_VARS    256
#define N_MARK    4
#define N_TOK     260          // N_VARS + N_MARK
#define D_MODEL   256
#define D_INNER   512
#define D_STATE   16
#define D_CONV    4
#define DT_RANK   16
#define PRED_LEN  96
#define M_ROWS    (BATCH * N_TOK)   // 8320 = 65 * 128

typedef __attribute__((ext_vector_type(16))) __bf16 v16bf;
typedef __attribute__((ext_vector_type(8)))  __bf16 v8bf;
typedef __attribute__((ext_vector_type(8)))  float  v8f;

union AFrag { v16bf v; v8bf h[2]; };

__device__ __forceinline__ v8f v8f_zero() {
    v8f z = {0.f,0.f,0.f,0.f,0.f,0.f,0.f,0.f};
    return z;
}

// A fragment: 16x32 bf16 tile of row-major X[M,K] at (m0,k0).
// lane L: row = m0 + (L&15); half = L>>4.
// v16bf elems e = K[k0 + half*16 + e]
__device__ __forceinline__ v16bf load_a_frag(const __bf16* __restrict__ arow,
                                             int k0, int half) {
    const __bf16* p = arow + k0 + half * 16;
    AFrag f;
    f.v = *(const v16bf*)(p);
    return f.v;
}

// B fragment: 32x16 tile B(k,n) = W[n][k], W row-major [N,K], tile at (k0,n0).
// lane L: col = n0 + (L&15); half = L>>4; elems e: K = k0 + half*16 + e.
__device__ __forceinline__ v16bf load_b_frag(const __bf16* __restrict__ W,
                                             int K, int n0, int k0, int lane) {
    const int col  = n0 + (lane & 15);
    const int half = lane >> 4;
    return *(const v16bf*)(W + (size_t)col * K + k0 + half * 16);
}

// NSUB: 16-col subtiles per wave (branch-free; grid.y * NSUB * 16 == N exactly)
// EPI: 0 = none, 1 = +bias, 2 = gelu(x+bias) exact, 3 = softplus(x+bias)
template<int NSUB, int EPI, bool OUT_F32, bool OUT_BF16>
__global__ __launch_bounds__(256)
void k_gemm_wmma(const __bf16* __restrict__ X, const __bf16* __restrict__ W,
                 const float* __restrict__ bias,
                 float* __restrict__ outF, __bf16* __restrict__ outB,
                 int N, int K)
{
    const int lane  = threadIdx.x & 31;
    const int wave  = threadIdx.x >> 5;
    const int m0    = (blockIdx.x * 8 + wave) * 16;
    const int nbase = blockIdx.y * (NSUB * 16);
    const int half  = lane >> 4;

    v8f acc[NSUB];
    #pragma unroll
    for (int j = 0; j < NSUB; ++j) acc[j] = v8f_zero();

    const __bf16* arow = X + (size_t)(m0 + (lane & 15)) * K;

    for (int k0 = 0; k0 < K; k0 += 32) {
        v16bf a = load_a_frag(arow, k0, half);
        __builtin_prefetch(arow + k0 + 32, 0, 0);   // speculative; OOB dropped
        #pragma unroll
        for (int j = 0; j < NSUB; ++j) {
            v16bf b = load_b_frag(W, K, nbase + j * 16, k0, lane);
            acc[j] = __builtin_amdgcn_wmma_f32_16x16x32_bf16(
                false, a, false, b, (short)0, acc[j], false, false);
        }
    }

    const int col_l = lane & 15;
    const int rowb  = m0 + half * 8;               // C/D: M = half*8 + r
    #pragma unroll
    for (int j = 0; j < NSUB; ++j) {
        const int col = nbase + j * 16 + col_l;
        const float bv = (EPI != 0) ? bias[col] : 0.f;
        #pragma unroll
        for (int r = 0; r < 8; ++r) {
            float v = acc[j][r];
            if (EPI == 1) { v += bv; }
            else if (EPI == 2) { v += bv; v = 0.5f * v * (1.f + erff(v * 0.70710678118654752f)); }
            else if (EPI == 3) { v += bv; v = (v > 20.f) ? v : log1pf(__expf(v)); }
            const size_t idx = (size_t)(rowb + r) * N + col;
            if (OUT_F32)  outF[idx] = v;
            if (OUT_BF16) outB[idx] = (__bf16)v;
        }
    }
}

// ---------------- elementwise / prep kernels ----------------

// per-(b,n) mean & std over time (coalesced across n)
__global__ void k_stats(const float* __restrict__ x,
                        float* __restrict__ mean, float* __restrict__ stdv)
{
    const int i = blockIdx.x * blockDim.x + threadIdx.x;
    if (i >= BATCH * N_VARS) return;
    const int n = i & (N_VARS - 1);
    const int b = i >> 8;
    float s = 0.f, s2 = 0.f;
    for (int t = 0; t < SEQ_LEN; ++t) {
        const float v = x[((size_t)b * SEQ_LEN + t) * N_VARS + n];
        s += v; s2 += v * v;
    }
    const float m   = s * (1.f / SEQ_LEN);
    const float var = s2 * (1.f / SEQ_LEN) - m * m;
    mean[i] = m;
    stdv[i] = sqrtf(var + 1e-5f);
}

// tokens[b][v][t]: normalized variates + raw mark variates, as bf16
__global__ void k_tokens(const float* __restrict__ xenc, const float* __restrict__ xmark,
                         const float* __restrict__ mean, const float* __restrict__ stdv,
                         __bf16* __restrict__ tok)
{
    const int i = blockIdx.x * blockDim.x + threadIdx.x;   // = (b*260+v)*512+t
    if (i >= BATCH * N_TOK * SEQ_LEN) return;
    const int t = i & (SEQ_LEN - 1);
    const int v = (i >> 9) % N_TOK;
    const int b = i / (N_TOK * SEQ_LEN);
    float val;
    if (v < N_VARS) {
        const int bn = b * N_VARS + v;
        val = (xenc[((size_t)b * SEQ_LEN + t) * N_VARS + v] - mean[bn]) / stdv[bn];
    } else {
        val = xmark[((size_t)b * SEQ_LEN + t) * N_MARK + (v - N_VARS)];
    }
    tok[i] = (__bf16)val;
}

__global__ void k_cvt_bf16(const float* __restrict__ src, __bf16* __restrict__ dst, int n)
{
    const int i = blockIdx.x * blockDim.x + threadIdx.x;
    if (i < n) dst[i] = (__bf16)src[i];
}

// pad a [rows, src_stride] f32 source (first 16 cols used) into [rows, 32] bf16
__global__ void k_pad32(const float* __restrict__ src, int src_stride,
                        __bf16* __restrict__ dst, int rows)
{
    const int i = blockIdx.x * blockDim.x + threadIdx.x;
    if (i >= rows * 32) return;
    const int c = i & 31, r = i >> 5;
    dst[i] = (c < DT_RANK) ? (__bf16)src[(size_t)r * src_stride + c] : (__bf16)0.f;
}

// causal depthwise conv (kernel 4) over token dim + SiLU; xm = xz[:, :512]
__global__ void k_conv_silu(const float* __restrict__ xz, const float* __restrict__ Wc,
                            const float* __restrict__ bc,
                            float* __restrict__ xmF, __bf16* __restrict__ xmB)
{
    const int i = blockIdx.x * blockDim.x + threadIdx.x;   // over M_ROWS*512
    if (i >= M_ROWS * D_INNER) return;
    const int d   = i & (D_INNER - 1);
    const int row = i >> 9;
    const int v   = row % N_TOK;
    const int b   = row / N_TOK;
    float acc = bc[d];
    #pragma unroll
    for (int k = 0; k < D_CONV; ++k) {
        const int vp = v - (D_CONV - 1) + k;
        if (vp >= 0)
            acc += Wc[d * D_CONV + k] * xz[(size_t)(b * N_TOK + vp) * (2 * D_INNER) + d];
    }
    const float s = acc / (1.f + __expf(-acc));
    xmF[i] = s;
    xmB[i] = (__bf16)s;
}

// selective scan: one block per batch, thread = channel d; 260 serial steps.
// Fuses D-skip and SiLU(z) gating; writes bf16 for the out-proj GEMM.
__global__ __launch_bounds__(D_INNER)
void k_scan(const float* __restrict__ dt, const float* __restrict__ xdbl,
            const float* __restrict__ xm, const float* __restrict__ xz,
            const float* __restrict__ A_log, const float* __restrict__ Dp,
            __bf16* __restrict__ ygate)
{
    const int b = blockIdx.x;
    const int d = threadIdx.x;
    float A[D_STATE], h[D_STATE];
    #pragma unroll
    for (int s = 0; s < D_STATE; ++s) {
        A[s] = -__expf(A_log[d * D_STATE + s]);
        h[s] = 0.f;
    }
    const float Dd = Dp[d];
    __shared__ float sB[D_STATE], sC[D_STATE];

    for (int v = 0; v < N_TOK; ++v) {
        const int row = b * N_TOK + v;
        __syncthreads();
        if (threadIdx.x < 32) {   // cols 16..47 of x_dbl: Bm then Cm
            const float val = xdbl[(size_t)row * 48 + 16 + threadIdx.x];
            if (threadIdx.x < 16) sB[threadIdx.x] = val;
            else                  sC[threadIdx.x - 16] = val;
        }
        __syncthreads();
        const float dtv = dt[(size_t)row * D_INNER + d];
        const float xv  = xm[(size_t)row * D_INNER + d];
        float y = 0.f;
        #pragma unroll
        for (int s = 0; s < D_STATE; ++s) {
            const float dA = __expf(dtv * A[s]);
            h[s] = dA * h[s] + (dtv * sB[s]) * xv;
            y += h[s] * sC[s];
        }
        const float z  = xz[(size_t)row * (2 * D_INNER) + D_INNER + d];
        const float sz = z / (1.f + __expf(-z));
        ygate[(size_t)row * D_INNER + d] = (__bf16)((y + Dd * xv) * sz);
    }
}

// de-normalize: out[b,t,n] = proj[b,n,t] * std[b,n] + mean[b,n]
__global__ void k_final(const float* __restrict__ proj, const float* __restrict__ mean,
                        const float* __restrict__ stdv, float* __restrict__ out)
{
    const int i = blockIdx.x * blockDim.x + threadIdx.x;   // = (b*96+t)*256+n
    if (i >= BATCH * PRED_LEN * N_VARS) return;
    const int n = i & (N_VARS - 1);
    const int t = (i >> 8) % PRED_LEN;
    const int b = i / (PRED_LEN * N_VARS);
    const int bn = b * N_VARS + n;
    out[i] = proj[((size_t)(b * N_TOK + n)) * PRED_LEN + t] * stdv[bn] + mean[bn];
}

// ---------------- host side ----------------
static inline int ceil_div(int a, int b) { return (a + b - 1) / b; }

extern "C" void kernel_launch(void* const* d_in, const int* in_sizes, int n_in,
                              void* d_out, int out_size, void* d_ws, size_t ws_size,
                              hipStream_t stream)
{
    const float* x_enc   = (const float*)d_in[0];
    const float* x_mark  = (const float*)d_in[1];
    const float* W_embed = (const float*)d_in[4];
    const float* b_embed = (const float*)d_in[5];
    const float* W_in    = (const float*)d_in[6];
    const float* W_conv  = (const float*)d_in[7];
    const float* b_conv  = (const float*)d_in[8];
    const float* W_xproj = (const float*)d_in[9];
    const float* W_dt    = (const float*)d_in[10];
    const float* b_dt    = (const float*)d_in[11];
    const float* A_log   = (const float*)d_in[12];
    const float* Dp      = (const float*)d_in[13];
    const float* W_out   = (const float*)d_in[14];
    const float* W_p1    = (const float*)d_in[15];
    const float* b_p1    = (const float*)d_in[16];
    const float* W_p2    = (const float*)d_in[17];
    const float* b_p2    = (const float*)d_in[18];

    // workspace carve-out (256B aligned)
    size_t off = 0;
    auto alloc = [&](size_t bytes) -> void* {
        void* p = (char*)d_ws + off;
        off = (off + bytes + 255) & ~(size_t)255;
        return p;
    };
    const int M = M_ROWS;
    float*  mean   = (float*)alloc(BATCH * N_VARS * 4);
    float*  stdv   = (float*)alloc(BATCH * N_VARS * 4);
    __bf16* tok    = (__bf16*)alloc((size_t)M * SEQ_LEN * 2);
    __bf16* Wemb_b = (__bf16*)alloc((size_t)D_MODEL * SEQ_LEN * 2);
    __bf16* x1     = (__bf16*)alloc((size_t)M * D_MODEL * 2);
    __bf16* Win_b  = (__bf16*)alloc((size_t)2 * D_INNER * D_MODEL * 2);
    float*  xz     = (float*)alloc((size_t)M * 2 * D_INNER * 4);
    float*  xm_f   = (float*)alloc((size_t)M * D_INNER * 4);
    __bf16* xm_b   = (__bf16*)alloc((size_t)M * D_INNER * 2);
    __bf16* Wxp_b  = (__bf16*)alloc((size_t)48 * D_INNER * 2);
    float*  xdbl   = (float*)alloc((size_t)M * 48 * 4);
    __bf16* Wdt_b  = (__bf16*)alloc((size_t)D_INNER * 32 * 2);
    __bf16* dtpad  = (__bf16*)alloc((size_t)M * 32 * 2);
    float*  dt     = (float*)alloc((size_t)M * D_INNER * 4);
    __bf16* ygate  = (__bf16*)alloc((size_t)M * D_INNER * 2);
    __bf16* Wout_b = (__bf16*)alloc((size_t)D_MODEL * D_INNER * 2);
    __bf16* x2     = (__bf16*)alloc((size_t)M * D_MODEL * 2);
    __bf16* Wp1_b  = (__bf16*)alloc((size_t)2 * D_MODEL * D_MODEL * 2);
    __bf16* h1     = (__bf16*)alloc((size_t)M * 2 * D_MODEL * 2);
    __bf16* Wp2_b  = (__bf16*)alloc((size_t)PRED_LEN * 2 * D_MODEL * 2);
    float*  proj   = (float*)alloc((size_t)M * PRED_LEN * 4);
    (void)ws_size;

    const int TB = 256;

    // 1) instance-norm stats
    k_stats<<<ceil_div(BATCH * N_VARS, TB), TB, 0, stream>>>(x_enc, mean, stdv);
    // 2) build bf16 token matrix [M, 512]
    k_tokens<<<ceil_div(BATCH * N_TOK * SEQ_LEN, TB), TB, 0, stream>>>(x_enc, x_mark, mean, stdv, tok);
    // 3) weight conversions
    k_cvt_bf16<<<ceil_div(D_MODEL * SEQ_LEN, TB), TB, 0, stream>>>(W_embed, Wemb_b, D_MODEL * SEQ_LEN);
    k_cvt_bf16<<<ceil_div(2 * D_INNER * D_MODEL, TB), TB, 0, stream>>>(W_in, Win_b, 2 * D_INNER * D_MODEL);
    k_cvt_bf16<<<ceil_div(48 * D_INNER, TB), TB, 0, stream>>>(W_xproj, Wxp_b, 48 * D_INNER);
    k_cvt_bf16<<<ceil_div(D_MODEL * D_INNER, TB), TB, 0, stream>>>(W_out, Wout_b, D_MODEL * D_INNER);
    k_cvt_bf16<<<ceil_div(2 * D_MODEL * D_MODEL, TB), TB, 0, stream>>>(W_p1, Wp1_b, 2 * D_MODEL * D_MODEL);
    k_cvt_bf16<<<ceil_div(PRED_LEN * 2 * D_MODEL, TB), TB, 0, stream>>>(W_p2, Wp2_b, PRED_LEN * 2 * D_MODEL);
    k_pad32<<<ceil_div(D_INNER * 32, TB), TB, 0, stream>>>(W_dt, DT_RANK, Wdt_b, D_INNER);

    // 4) embed: x1 = tok @ W_embed^T + b_embed  (bf16 out)   N=256=4*64
    {   dim3 g(M / 128, D_MODEL / 64);
        k_gemm_wmma<4, 1, false, true><<<g, TB, 0, stream>>>(tok, Wemb_b, b_embed, nullptr, x1, D_MODEL, SEQ_LEN); }
    // 5) in-proj: xz = x1 @ W_in^T  (f32 out)   N=1024=16*64
    {   dim3 g(M / 128, (2 * D_INNER) / 64);
        k_gemm_wmma<4, 0, true, false><<<g, TB, 0, stream>>>(x1, Win_b, nullptr, xz, nullptr, 2 * D_INNER, D_MODEL); }
    // 6) causal depthwise conv + SiLU
    k_conv_silu<<<ceil_div(M * D_INNER, TB), TB, 0, stream>>>(xz, W_conv, b_conv, xm_f, xm_b);
    // 7) x-proj: xdbl = xm @ W_xproj^T  (f32 out)   N=48=1*48 -> NSUB=3
    {   dim3 g(M / 128, 1);
        k_gemm_wmma<3, 0, true, false><<<g, TB, 0, stream>>>(xm_b, Wxp_b, nullptr, xdbl, nullptr, 48, D_INNER); }
    // 8) pad dt slice to K=32 bf16
    k_pad32<<<ceil_div(M * 32, TB), TB, 0, stream>>>(xdbl, 48, dtpad, M);
    // 9) dt-proj: dt = softplus(dtpad @ Wdt^T + b_dt)  (f32 out)   N=512=8*64
    {   dim3 g(M / 128, D_INNER / 64);
        k_gemm_wmma<4, 3, true, false><<<g, TB, 0, stream>>>(dtpad, Wdt_b, b_dt, dt, nullptr, D_INNER, 32); }
    // 10) selective scan + D-skip + SiLU gate (bf16 out)
    k_scan<<<BATCH, D_INNER, 0, stream>>>(dt, xdbl, xm_f, xz, A_log, Dp, ygate);
    // 11) out-proj: x2 = ygate @ W_out^T  (bf16 out)   N=256=4*64
    {   dim3 g(M / 128, D_MODEL / 64);
        k_gemm_wmma<4, 0, false, true><<<g, TB, 0, stream>>>(ygate, Wout_b, nullptr, nullptr, x2, D_MODEL, D_INNER); }
    // 12) head p1: h1 = gelu(x2 @ W_p1^T + b_p1)  (bf16 out)   N=512=8*64
    {   dim3 g(M / 128, (2 * D_MODEL) / 64);
        k_gemm_wmma<4, 2, false, true><<<g, TB, 0, stream>>>(x2, Wp1_b, b_p1, nullptr, h1, 2 * D_MODEL, D_MODEL); }
    // 13) head p2: proj = h1 @ W_p2^T + b_p2  (f32 out)   N=96=2*48 -> NSUB=3, gy=2
    {   dim3 g(M / 128, 2);
        k_gemm_wmma<3, 1, true, false><<<g, TB, 0, stream>>>(h1, Wp2_b, b_p2, proj, nullptr, PRED_LEN, 2 * D_MODEL); }
    // 14) de-normalize into output [B, 96, 256]
    k_final<<<ceil_div(BATCH * PRED_LEN * N_VARS, TB), TB, 0, stream>>>(proj, mean, stdv, (float*)d_out);
}
